// TwoAxisTransformerEncoderLayer_47296179863598
// MI455X (gfx1250) — matrix operand order repaired and checked by reading, not compiled, not run on hardware
//
#include <hip/hip_runtime.h>

// ---------------------------------------------------------------------------
// TwoAxisTransformerEncoderLayer for MI455X (gfx1250), wave32 + WMMA bf16.
// All GEMMs / attention matmuls use v_wmma_f32_16x16x32_bf16 (fp32 accum).
// GEMM tiles staged into LDS via global_load_async_to_lds_b128 (the gfx1250
// async LDS-DMA path, ASYNCcnt-tracked); guarded fallback to vector loads.
// Shapes (from setup_inputs, static): bs=8, R=128, C=64, E=256, H=8, hd=32,
// S=96 (ctx/qry split, python scalar -> compiled in), MLP hidden=1024.
// ---------------------------------------------------------------------------

typedef __attribute__((ext_vector_type(16))) __bf16 v16bf;
typedef __attribute__((ext_vector_type(8)))  float  v8f;
typedef unsigned short u16;
typedef unsigned long long uptr;
typedef int v4i __attribute__((vector_size(16)));   // param type of the async builtin

union FragAB { v16bf v; u16 u[16]; };

#if defined(__has_builtin)
#if __has_builtin(__builtin_amdgcn_global_load_async_to_lds_b128)
#define USE_ASYNC_LDS 1
#endif
#endif
#ifndef USE_ASYNC_LDS
#define USE_ASYNC_LDS 0
#endif

#define AS1 __attribute__((address_space(1)))
#define AS3 __attribute__((address_space(3)))

// Copy 16 bytes global -> LDS. Async path uses the gfx1250 LDS-DMA
// (ASYNCcnt-tracked, no VGPR round trip); fallback is a b128 load/store.
__device__ __forceinline__ void copy16_g2l(const u16* __restrict__ g, u16* l) {
#if USE_ASYNC_LDS
    __builtin_amdgcn_global_load_async_to_lds_b128(
        (AS1 v4i*)(uptr)g,                 // global source (addrspace(1) int4*)
        (AS3 v4i*)(unsigned)(uptr)l,       // LDS dest: low 32 bits of flat LDS addr == LDS offset
        0, 0);                             // imm offset, cpol
#else
    *(uint4*)l = *(const uint4*)g;
#endif
}

__device__ __forceinline__ void wait_async_lds() {
#if USE_ASYNC_LDS
#if __has_builtin(__builtin_amdgcn_s_wait_asynccnt)
    __builtin_amdgcn_s_wait_asynccnt(0);
#else
    asm volatile("s_wait_asynccnt 0x0" ::: "memory");
#endif
#endif
}

__device__ __forceinline__ u16 f32_to_bf16(float f) {
    unsigned int u = __float_as_uint(f);
    unsigned int r = u + 0x7FFFu + ((u >> 16) & 1u);
    return (u16)(r >> 16);
}

__device__ __forceinline__ float clean1(float v) {
    if (v != v) return 0.0f;
    if (v == __builtin_inff() || v == -__builtin_inff()) return 0.0f;
    return v;
}

// ------------------------------ elementwise --------------------------------

__global__ void k_clean(const float* __restrict__ in, float* __restrict__ out, long n) {
    long i = blockIdx.x * (long)blockDim.x + threadIdx.x;
    if (i < n) out[i] = clean1(in[i]);
}

__global__ void k_cvt(const float* __restrict__ in, u16* __restrict__ out, long n) {
    long i = blockIdx.x * (long)blockDim.x + threadIdx.x;
    if (i < n) out[i] = f32_to_bf16(in[i]);
}

// compact (j over nrows, Sloc per seq) -> gather from full layout row
__global__ void k_cvt_gather(const float* __restrict__ in, u16* __restrict__ out,
                             long nrows, int Sloc, int Rfull, int tokOff) {
    long idx = blockIdx.x * (long)blockDim.x + threadIdx.x;
    long total = nrows * 256;
    if (idx >= total) return;
    int  e = (int)(idx & 255);
    long j = idx >> 8;
    long full = (j / Sloc) * Rfull + tokOff + (j % Sloc);
    out[idx] = f32_to_bf16(in[full * 256 + e]);
}

// out[full] = comp[j] + resid[full]
__global__ void k_add_scatter(const float* __restrict__ comp, const float* __restrict__ resid,
                              float* __restrict__ out,
                              long nrows, int Sloc, int Rfull, int tokOff) {
    long idx = blockIdx.x * (long)blockDim.x + threadIdx.x;
    long total = nrows * 256;
    if (idx >= total) return;
    int  e = (int)(idx & 255);
    long j = idx >> 8;
    long full = (j / Sloc) * Rfull + tokOff + (j % Sloc);
    out[full * 256 + e] = comp[idx] + resid[full * 256 + e];
}

// in (bs, X, Y, E) -> out (bs, Y, X, E), optional nan_to_num
__global__ void k_transpose(const float* __restrict__ in, float* __restrict__ out,
                            int bs, int X, int Y, int EE, int clean) {
    long idx = blockIdx.x * (long)blockDim.x + threadIdx.x;
    long total = (long)bs * X * Y * EE;
    if (idx >= total) return;
    int  e = (int)(idx % EE);
    long t = idx / EE;
    int  y = (int)(t % Y); t /= Y;
    int  x = (int)(t % X); t /= X;
    int  b = (int)t;
    float v = in[idx];
    if (clean) v = clean1(v);
    out[(((long)b * Y + y) * X + x) * EE + e] = v;
}

// RoPE on q (cols [0,256)) and/or k (cols [256,512)) regions of a (rows x 768)
// qkv buffer. pos = position within sequence of length Rfull (or within the
// compact ctx window when Sremap > 0).
__global__ void k_rope(float* __restrict__ qkv, long ntok,
                       int Sremap, int Rfull, int doQ, int doK) {
    long idx = blockIdx.x * (long)blockDim.x + threadIdx.x;
    long total = ntok * 256;   // ntok * H(8) * 16 pairs * 2 regions
    if (idx >= total) return;
    int  i  = (int)(idx & 15);
    int  h  = (int)((idx >> 4) & 7);
    int  qk = (int)((idx >> 7) & 1);
    long j  = idx >> 8;
    if (qk == 0 && !doQ) return;
    if (qk == 1 && !doK) return;
    long row; int pos;
    if (Sremap > 0) { row = (j / Sremap) * Rfull + (j % Sremap); pos = (int)(j % Sremap); }
    else            { row = j;                                   pos = (int)(j % Rfull); }
    float invf = __powf(10000.0f, -(float)(2 * i) / 32.0f);
    float ang  = (float)pos * invf;
    float c = __cosf(ang), s = __sinf(ang);
    long base = row * 768 + (long)qk * 256 + h * 32 + 2 * i;
    float xe = qkv[base], xo = qkv[base + 1];
    qkv[base]     = xe * c - xo * s;
    qkv[base + 1] = xe * s + xo * c;
}

// LayerNorm over E=256; one wave (32 lanes) per token, 8 tokens per block.
__launch_bounds__(256)
__global__ void k_ln(const float* __restrict__ in, const float* __restrict__ g,
                     const float* __restrict__ b, float* __restrict__ out, long ntok) {
    long tok  = blockIdx.x * 8L + (threadIdx.x >> 5);
    int  lane = threadIdx.x & 31;
    if (tok >= ntok) return;
    const float* row = in + tok * 256;
    float vals[8]; float s = 0.0f;
#pragma unroll
    for (int j = 0; j < 8; ++j) { vals[j] = row[lane + j * 32]; s += vals[j]; }
#pragma unroll
    for (int o = 16; o > 0; o >>= 1) s += __shfl_xor(s, o, 32);
    float mean = s * (1.0f / 256.0f);
    float vs = 0.0f;
#pragma unroll
    for (int j = 0; j < 8; ++j) { float d = vals[j] - mean; vs += d * d; }
#pragma unroll
    for (int o = 16; o > 0; o >>= 1) vs += __shfl_xor(vs, o, 32);
    float inv = rsqrtf(vs * (1.0f / 256.0f) + 1e-5f);
#pragma unroll
    for (int j = 0; j < 8; ++j) {
        int e = lane + j * 32;
        out[tok * 256 + e] = (vals[j] - mean) * inv * g[e] + b[e];
    }
}

// ------------------------------ WMMA GEMM ----------------------------------
// C[M,N] = A[M,K] @ W[N,K]^T (+bias) (+resid) (+GELU), bf16 in, f32 accum.
// Workgroup: 256 threads = 8 waves, 128x64 output tile, BK=32.
// Wave w: 32x32 register tile (wr=w&3 row slot, wc=w>>2 col slot):
// 2 A-frags x 2 B-frags -> 4 wmma per K-step (2x the wmma density / LDS byte
// of a 16x32 wave tile). Optional out-row remap scatters compact ctx rows.
__launch_bounds__(256)
__global__ void k_gemm(const u16* __restrict__ A, const u16* __restrict__ W,
                       const float* __restrict__ bias, const float* __restrict__ resid,
                       float* __restrict__ outF, u16* __restrict__ outH,
                       int M, int N, int K, int gelu, int remapS, int remapR) {
    __shared__ __align__(16) u16 lA[128 * 32];
    __shared__ __align__(16) u16 lB[64 * 32];
    const int tid  = threadIdx.x;
    const int lane = tid & 31, wave = tid >> 5;
    const int m0 = blockIdx.y * 128, n0 = blockIdx.x * 64;
    const int wr = wave & 3, wc = wave >> 2;
    const int half = lane >> 4, l16 = lane & 15;

    v8f acc00 = {}, acc01 = {}, acc10 = {}, acc11 = {};

    for (int k0 = 0; k0 < K; k0 += 32) {
        // stage 192 rows x 32 bf16 (A:128 rows, W:64 rows): 768 16B chunks, 3/thread
#pragma unroll
        for (int t = 0; t < 3; ++t) {
            int chunk = tid + 256 * t;
            int r = chunk >> 2;
            int c = (chunk & 3) * 8;
            if (r < 128) {
                copy16_g2l(A + (long)(m0 + r) * K + k0 + c, &lA[r * 32 + c]);
            } else {
                int rb = r - 128;
                copy16_g2l(W + (long)(n0 + rb) * K + k0 + c, &lB[rb * 32 + c]);
            }
        }
#if !USE_ASYNC_LDS
        if (k0 + 32 < K) {
            __builtin_prefetch(A + (long)(m0 + (tid >> 2)) * K + k0 + 32, 0, 3);
            __builtin_prefetch(W + (long)(n0 + (tid >> 2) - 128) * K + k0 + 32, 0, 3);
        }
#endif
        wait_async_lds();
        __syncthreads();

        FragAB a0, a1, b0, b1;
#pragma unroll
        for (int v = 0; v < 8; ++v) {
            // A (16x32, MxK): lanes 0-15 K={0..7,16..23}, lanes 16-31 K={8..15,24..31}
            int ka = (v < 4) ? (2 * v + 8 * half) : (16 + 2 * (v - 4) + 8 * half);
            a0.u[2 * v]     = lA[(wr * 32 + l16) * 32 + ka];
            a0.u[2 * v + 1] = lA[(wr * 32 + l16) * 32 + ka + 1];
            a1.u[2 * v]     = lA[(wr * 32 + 16 + l16) * 32 + ka];
            a1.u[2 * v + 1] = lA[(wr * 32 + 16 + l16) * 32 + ka + 1];
            // B (32x16, KxN): lanes 0-15 K={0..7,16..23}, lanes 16-31 K={8..15,24..31}, col = lane&15
            int kb = (v < 4) ? (2 * v + 8 * half) : (16 + 2 * (v - 4) + 8 * half);
            b0.u[2 * v]     = lB[(wc * 32 + l16) * 32 + kb];
            b0.u[2 * v + 1] = lB[(wc * 32 + l16) * 32 + kb + 1];
            b1.u[2 * v]     = lB[(wc * 32 + 16 + l16) * 32 + kb];
            b1.u[2 * v + 1] = lB[(wc * 32 + 16 + l16) * 32 + kb + 1];
        }
        acc00 = __builtin_amdgcn_wmma_f32_16x16x32_bf16(false, a0.v, false, b0.v, (short)0, acc00, false, false);
        acc01 = __builtin_amdgcn_wmma_f32_16x16x32_bf16(false, a0.v, false, b1.v, (short)0, acc01, false, false);
        acc10 = __builtin_amdgcn_wmma_f32_16x16x32_bf16(false, a1.v, false, b0.v, (short)0, acc10, false, false);
        acc11 = __builtin_amdgcn_wmma_f32_16x16x32_bf16(false, a1.v, false, b1.v, (short)0, acc11, false, false);
        __syncthreads();
    }

#pragma unroll
    for (int ia = 0; ia < 2; ++ia) {
#pragma unroll
        for (int i = 0; i < 8; ++i) {
            int m = m0 + wr * 32 + ia * 16 + i + 8 * half;  // C/D layout: lanes0-15 M=i, lanes16-31 M=8+i
            long orow = (remapS > 0) ? ((long)(m / remapS) * remapR + (m % remapS)) : (long)m;
            for (int ib = 0; ib < 2; ++ib) {
                int n = n0 + wc * 32 + ib * 16 + l16;
                float v = ia ? (ib ? acc11[i] : acc10[i]) : (ib ? acc01[i] : acc00[i]);
                if (bias)  v += bias[n];
                if (resid) v += resid[orow * N + n];
                if (gelu)  v = 0.5f * v * (1.0f + erff(v * 0.70710678118f));
                if (outF) outF[orow * N + n] = v;
                if (outH) outH[orow * N + n] = f32_to_bf16(v);
            }
        }
    }
}

// ---------------------------- WMMA attention -------------------------------
// One workgroup = 64 threads (2 waves) handles 32 q rows of one (seq, head).
// QK^T and P@V both via v_wmma_f32_16x16x32_bf16 (K = head_dim = 32).
// q/k/v read from fp32 qkv buffers (row stride 768: [q|k|v] each E=256 wide).
// Output compact: row = seq*Lq + qpos, stride 256, cols [h*32, h*32+32).
#define LKMAX 96
__launch_bounds__(64)
__global__ void k_attn(const float* __restrict__ qkvQ, const float* __restrict__ qkvKV,
                       const int* __restrict__ mask, float* __restrict__ out,
                       int seqStrideQ, int qTokOff, int seqStrideKV, int kvTokOff,
                       int Lq, int Lk, int maskDiv, int maskStride) {
    __shared__ __align__(16) u16 Qs[32 * 32];
    __shared__ __align__(16) u16 Ks[LKMAX * 32];
    __shared__ __align__(16) u16 Vs[LKMAX * 32];
    __shared__ float Sb[32 * LKMAX];
    __shared__ u16   Pb[32 * LKMAX];

    const int tid = threadIdx.x, lane = tid & 31, wave = tid >> 5;
    const int qt = blockIdx.x, h = blockIdx.y, seq = blockIdx.z;
    const int half = lane >> 4, l16 = lane & 15;
    const float scale = 0.17677669529f;   // 1/sqrt(32)

    for (int idx = tid; idx < 32 * 32; idx += 64) {
        int r = idx >> 5, d = idx & 31;
        long tok = (long)seq * seqStrideQ + qTokOff + qt * 32 + r;
        Qs[idx] = f32_to_bf16(qkvQ[tok * 768 + h * 32 + d] * scale);
    }
    for (int idx = tid; idx < Lk * 32; idx += 64) {
        int r = idx >> 5, d = idx & 31;
        long tok = (long)seq * seqStrideKV + kvTokOff + r;
        Ks[idx] = f32_to_bf16(qkvKV[tok * 768 + 256 + h * 32 + d]);
        Vs[idx] = f32_to_bf16(qkvKV[tok * 768 + 512 + h * 32 + d]);
    }
    __syncthreads();

    // --- S = Q K^T (per wave: 16 q rows x Lk) ---
    FragAB qf;
    const int qr = wave * 16 + l16;
#pragma unroll
    for (int v = 0; v < 8; ++v) {
        int ka = (v < 4) ? (2 * v + 8 * half) : (16 + 2 * (v - 4) + 8 * half);
        qf.u[2 * v]     = Qs[qr * 32 + ka];
        qf.u[2 * v + 1] = Qs[qr * 32 + ka + 1];
    }
    for (int kt = 0; kt < Lk / 16; ++kt) {
        FragAB kf;
#pragma unroll
        for (int v = 0; v < 8; ++v) {
            int kb = (v < 4) ? (2 * v + 8 * half) : (16 + 2 * (v - 4) + 8 * half);
            kf.u[2 * v]     = Ks[(kt * 16 + l16) * 32 + kb];
            kf.u[2 * v + 1] = Ks[(kt * 16 + l16) * 32 + kb + 1];
        }
        v8f s = {};
        s = __builtin_amdgcn_wmma_f32_16x16x32_bf16(false, qf.v, false, kf.v, (short)0, s, false, false);
        float biasv = 0.0f;
        if (maskDiv > 0) {
            int b = seq / maskDiv;
            int kidx = kt * 16 + l16;
            if (mask[b * maskStride + kidx] == 0) biasv = -1.0e9f;
        }
#pragma unroll
        for (int i = 0; i < 8; ++i) {
            int row = wave * 16 + i + 8 * half;
            Sb[row * LKMAX + kt * 16 + l16] = s[i] + biasv;
        }
    }
    __syncthreads();

    // --- softmax, one thread per q row ---
    if (tid < 32) {
        int row = tid;
        float m = -3.0e38f;
        for (int k = 0; k < Lk; ++k) m = fmaxf(m, Sb[row * LKMAX + k]);
        float sum = 0.0f;
        for (int k = 0; k < Lk; ++k) { float e = __expf(Sb[row * LKMAX + k] - m); Sb[row * LKMAX + k] = e; sum += e; }
        float inv = 1.0f / sum;
        for (int k = 0; k < Lk; ++k) Pb[row * LKMAX + k] = f32_to_bf16(Sb[row * LKMAX + k] * inv);
    }
    __syncthreads();

    // --- O = P V (K-loop over Lk in steps of 32) ---
    v8f o0 = {}, o1 = {};
    for (int kc = 0; kc < Lk / 32; ++kc) {
        FragAB pf, v0, v1;
#pragma unroll
        for (int v = 0; v < 8; ++v) {
            int ka = (v < 4) ? (2 * v + 8 * half) : (16 + 2 * (v - 4) + 8 * half);
            pf.u[2 * v]     = Pb[(wave * 16 + l16) * LKMAX + kc * 32 + ka];
            pf.u[2 * v + 1] = Pb[(wave * 16 + l16) * LKMAX + kc * 32 + ka + 1];
            int kb = (v < 4) ? (2 * v + 8 * half) : (16 + 2 * (v - 4) + 8 * half);
            v0.u[2 * v]     = Vs[(kc * 32 + kb) * 32 + l16];
            v0.u[2 * v + 1] = Vs[(kc * 32 + kb + 1) * 32 + l16];
            v1.u[2 * v]     = Vs[(kc * 32 + kb) * 32 + 16 + l16];
            v1.u[2 * v + 1] = Vs[(kc * 32 + kb + 1) * 32 + 16 + l16];
        }
        o0 = __builtin_amdgcn_wmma_f32_16x16x32_bf16(false, pf.v, false, v0.v, (short)0, o0, false, false);
        o1 = __builtin_amdgcn_wmma_f32_16x16x32_bf16(false, pf.v, false, v1.v, (short)0, o1, false, false);
    }
#pragma unroll
    for (int i = 0; i < 8; ++i) {
        int row = qt * 32 + wave * 16 + i + 8 * half;
        long orow = (long)seq * Lq + row;
        out[orow * 256 + h * 32 + l16]      = o0[i];
        out[orow * 256 + h * 32 + 16 + l16] = o1[i];
    }
}

// ------------------------------ orchestration ------------------------------

extern "C" void kernel_launch(void* const* d_in, const int* in_sizes, int n_in,
                              void* d_out, int out_size, void* d_ws, size_t ws_size,
                              hipStream_t stream) {
    constexpr long NTOK = 65536;          // 8*128*64
    constexpr long NE   = NTOK * 256;     // 16,777,216
    constexpr int  Sctx = 96, Rfull = 128, Cc = 64;
    constexpr long M2 = 512L * 96;        // 49152 ctx rows (bs*C * S)
    constexpr long M3 = 512L * 32;        // 16384 qry rows

    const float* src    = (const float*)d_in[0];
    const int*   mask   = (const int*)d_in[1];
    const float* y_in_w = (const float*)d_in[3];
    const float* y_in_b = (const float*)d_in[4];
    const float* y_out_w= (const float*)d_in[5];
    const float* y_out_b= (const float*)d_in[6];
    const float* x_in_w = (const float*)d_in[7];
    const float* x_in_b = (const float*)d_in[8];
    const float* x_out_w= (const float*)d_in[9];
    const float* x_out_b= (const float*)d_in[10];
    const float* w1  = (const float*)d_in[11];
    const float* b1  = (const float*)d_in[12];
    const float* w2  = (const float*)d_in[13];
    const float* b2  = (const float*)d_in[14];
    const float* n1g = (const float*)d_in[15]; const float* n1b = (const float*)d_in[16];
    const float* n2g = (const float*)d_in[17]; const float* n2b = (const float*)d_in[18];
    const float* n3g = (const float*)d_in[19]; const float* n3b = (const float*)d_in[20];

    char* ws = (char*)d_ws;
    size_t off = 0;
    auto carve = [&](size_t bytes) { char* p = ws + off; off += (bytes + 255) & ~(size_t)255; return p; };
    float* T0  = (float*)carve(NE * 4);            // activations (full layout)
    float* T1  = (float*)carve(NE * 4);
    float* T2  = (float*)carve(NE * 4);
    float* QKV = (float*)carve(NTOK * 768 * 4);    // also reused as bf16 MLP hidden
    u16*   XBF = (u16*)  carve(NE * 2);            // bf16 GEMM input staging
    u16* WYIN  = (u16*)carve(768 * 256 * 2);
    u16* WYOUT = (u16*)carve(256 * 256 * 2);
    u16* WXIN  = (u16*)carve(768 * 256 * 2);
    u16* WXOUT = (u16*)carve(256 * 256 * 2);
    u16* W1B   = (u16*)carve(1024 * 256 * 2);
    u16* W2B   = (u16*)carve(256 * 1024 * 2);
    u16* HBF   = (u16*)QKV;                        // bf16 hidden (N x 1024) aliases QKV

    const dim3 blk256(256), blk64(64);
    const unsigned gNE = (unsigned)(NE / 256);

    // ---- weight conversions (fp32 -> bf16) ----
    k_cvt<<<768,  blk256, 0, stream>>>(y_in_w,  WYIN,  768 * 256);
    k_cvt<<<256,  blk256, 0, stream>>>(y_out_w, WYOUT, 256 * 256);
    k_cvt<<<768,  blk256, 0, stream>>>(x_in_w,  WXIN,  768 * 256);
    k_cvt<<<256,  blk256, 0, stream>>>(x_out_w, WXOUT, 256 * 256);
    k_cvt<<<1024, blk256, 0, stream>>>(w1, W1B, 1024 * 256);
    k_cvt<<<1024, blk256, 0, stream>>>(w2, W2B, 256 * 1024);

    // ---- stage A: row attention (1024 seqs of len 64) ----
    k_clean<<<gNE, blk256, 0, stream>>>(src, T0, NE);
    k_cvt<<<gNE, blk256, 0, stream>>>(T0, XBF, NE);
    k_gemm<<<dim3(12, 512), blk256, 0, stream>>>(XBF, WYIN, y_in_b, nullptr, QKV, nullptr,
                                                 65536, 768, 256, 0, 0, 0);
    k_attn<<<dim3(2, 8, 1024), blk64, 0, stream>>>(QKV, QKV, nullptr, T1,
                                                   64, 0, 64, 0, 64, 64, 0, 0);
    k_cvt<<<gNE, blk256, 0, stream>>>(T1, XBF, NE);
    k_gemm<<<dim3(4, 512), blk256, 0, stream>>>(XBF, WYOUT, y_out_b, T0, T2, nullptr,
                                                65536, 256, 256, 0, 0, 0);
    k_ln<<<8192, blk256, 0, stream>>>(T2, n1g, n1b, T1, NTOK);

    // ---- transpose to (bs, C, R, E) with nan_to_num ----
    k_transpose<<<gNE, blk256, 0, stream>>>(T1, T0, 8, 128, 64, 256, 1);

    // ---- stage B: column attention, ctx pass ----
    k_cvt<<<gNE, blk256, 0, stream>>>(T0, XBF, NE);
    k_gemm<<<dim3(12, 512), blk256, 0, stream>>>(XBF, WXIN, x_in_b, nullptr, QKV, nullptr,
                                                 65536, 768, 256, 0, 0, 0);
    k_rope<<<(unsigned)NTOK, blk256, 0, stream>>>(QKV, NTOK, 0, Rfull, 1, 1);
    k_attn<<<dim3(3, 8, 512), blk64, 0, stream>>>(QKV, QKV, mask, T2,
                                                  Rfull, 0, Rfull, 0, 96, 96, Cc, Rfull);
    k_cvt<<<(unsigned)M2, blk256, 0, stream>>>(T2, XBF, M2 * 256);
    k_gemm<<<dim3(4, 384), blk256, 0, stream>>>(XBF, WXOUT, x_out_b, nullptr, T2, nullptr,
                                                (int)M2, 256, 256, 0, 0, 0);
    k_add_scatter<<<(unsigned)M2, blk256, 0, stream>>>(T2, T0, T1, M2, Sctx, Rfull, 0);

    // ---- recompute k,v of updated ctx (scatter new qkv into ctx rows) ----
    k_cvt_gather<<<(unsigned)M2, blk256, 0, stream>>>(T1, XBF, M2, Sctx, Rfull, 0);
    k_gemm<<<dim3(12, 384), blk256, 0, stream>>>(XBF, WXIN, x_in_b, nullptr, QKV, nullptr,
                                                 (int)M2, 768, 256, 0, Sctx, Rfull);
    k_rope<<<(unsigned)M2, blk256, 0, stream>>>(QKV, M2, Sctx, Rfull, 0, 1);

    // ---- stage C: column attention, qry pass ----
    k_attn<<<dim3(1, 8, 512), blk64, 0, stream>>>(QKV, QKV, mask, T2,
                                                  Rfull, Sctx, Rfull, 0, 32, 96, Cc, Rfull);
    k_cvt<<<(unsigned)M3, blk256, 0, stream>>>(T2, XBF, M3 * 256);
    k_gemm<<<dim3(4, 128), blk256, 0, stream>>>(XBF, WXOUT, x_out_b, nullptr, T2, nullptr,
                                                (int)M3, 256, 256, 0, 0, 0);
    k_add_scatter<<<(unsigned)M3, blk256, 0, stream>>>(T2, T0, T1, M3, 32, Rfull, Sctx);

    // ---- transpose back to (bs, R, C, E), LN2, MLP, LN3 ----
    k_transpose<<<gNE, blk256, 0, stream>>>(T1, T0, 8, 64, 128, 256, 0);
    k_ln<<<8192, blk256, 0, stream>>>(T0, n2g, n2b, T2, NTOK);
    k_cvt<<<gNE, blk256, 0, stream>>>(T2, XBF, NE);
    k_gemm<<<dim3(16, 512), blk256, 0, stream>>>(XBF, W1B, b1, nullptr, nullptr, HBF,
                                                 65536, 1024, 256, 1, 0, 0);      // GELU, bf16 out
    k_gemm<<<dim3(4, 512), blk256, 0, stream>>>(HBF, W2B, b2, T2, T1, nullptr,
                                                65536, 256, 1024, 0, 0, 0);       // + residual
    k_ln<<<8192, blk256, 0, stream>>>(T1, n3g, n3b, (float*)d_out, NTOK);
}
